// Attn_44143673868671
// MI455X (gfx1250) — matrix operand order repaired and checked by reading, not compiled, not run on hardware
//
#include <hip/hip_runtime.h>
#include <hip/hip_bf16.h>
#include <math.h>

// ---------------- problem constants ----------------
#define BB     2
#define TT     1024
#define CC     1024
#define HH     16
#define NOPE   32
#define ROPED  64
#define VDIM   32
#define QLORA  96
#define KVLORA 32
#define BLKSZ  16
#define WINSZ  256
#define KEEPN  256
#define NBLK   (TT / BLKSZ)      // 64
#define QKD    (NOPE + ROPED)    // 96
#define EPSF   1e-6f
#define SCALEF 0.10206207261596575f   // 1/sqrt(96)

typedef __bf16 bf16;
typedef __attribute__((ext_vector_type(16))) __bf16 v16bf;
typedef __attribute__((ext_vector_type(8)))  __bf16 v8bf;
typedef __attribute__((ext_vector_type(8)))  float  v8f;

// ---------------- WMMA fragment helpers ----------------
// A (16x32 bf16): lanes 0-15 hold K 0..15, lanes 16-31 hold K 16..31.
// B (32x16 bf16): lanes 0-15 hold K 0..15, lanes 16-31 hold K 16..31.
__device__ __forceinline__ v16bf cat8(v8bf lo, v8bf hi) {
  return __builtin_shufflevector(lo, hi, 0, 1, 2, 3, 4, 5, 6, 7, 8, 9, 10, 11, 12,
                                 13, 14, 15);
}
__device__ __forceinline__ v8f wmma_bf16(v16bf a, v16bf b, v8f c) {
  return __builtin_amdgcn_wmma_f32_16x16x32_bf16(false, a, false, b, (short)0, c,
                                                 false, false);
}

// ---------------- weight transform: fp32 [K,N] -> bf16 col-major [N,K] ------
__global__ void wtrans_kernel(const float* __restrict__ W, bf16* __restrict__ WT,
                              int K, int N) {
  __shared__ float tile[32][33];
  int kt = blockIdx.y * 32, nt = blockIdx.x * 32;
  int tx = threadIdx.x & 31, ty = threadIdx.x >> 5;  // 256 threads = 32x8
  for (int r = ty; r < 32; r += 8) {
    int k = kt + r, n = nt + tx;
    tile[r][tx] = (k < K && n < N) ? W[(long)k * N + n] : 0.f;
  }
  __syncthreads();
  for (int r = ty; r < 32; r += 8) {
    int n = nt + r, k = kt + tx;
    if (n < N && k < K) WT[(long)n * K + k] = (bf16)tile[tx][r];
  }
}

// ---------------- fp32 -> bf16 row-major cast -------------------------------
__global__ void castbf_kernel(const float* __restrict__ X, bf16* __restrict__ Y,
                              long n) {
  long i = blockIdx.x * (long)blockDim.x + threadIdx.x;
  if (i < n) Y[i] = (bf16)X[i];
}

// ---------------- generic WMMA GEMM: out = act(A[M,K] @ W[K,N]) -------------
// A: bf16 row-major. WT: bf16 col-major [N,K]. Optional fp32 / bf16 outputs.
// Grid tiles M,N exactly (M%64==0, N%(16*NT)==0): no bounds checks.
// Register double-buffered: loads for step k+32 issued before WMMAs of step k.
template <int NT>
__global__ void gemm_bf16_kernel(const bf16* __restrict__ A,
                                 const bf16* __restrict__ WT,
                                 float* __restrict__ OutF, bf16* __restrict__ OutB,
                                 int M, int N, int K, int act) {
  const int lane = threadIdx.x & 31;
  const int wave = threadIdx.x >> 5;
  const int m0 = (blockIdx.y * 4 + wave) * 16;
  const int n0 = blockIdx.x * NT * 16;
  const int hi = lane >> 4;
  const bf16* abase = A + (long)(m0 + (lane & 15)) * K + hi * 16;
  const bf16* wcol[NT];
#pragma unroll
  for (int t = 0; t < NT; ++t)
    wcol[t] = WT + (long)(n0 + t * 16 + (lane & 15)) * K + hi * 16;

  v8f acc[NT];
#pragma unroll
  for (int t = 0; t < NT; ++t) acc[t] = (v8f){};

  // prologue: fragment set for k0 = 0
  v16bf a_cur = cat8(*(const v8bf*)(abase), *(const v8bf*)(abase + 8));
  v16bf b_cur[NT];
#pragma unroll
  for (int t = 0; t < NT; ++t)
    b_cur[t] = cat8(*(const v8bf*)(wcol[t]), *(const v8bf*)(wcol[t] + 8));

  for (int k0 = 0; k0 + 32 < K; k0 += 32) {
    const int kn = k0 + 32;
    // issue next-step loads first (overlap with this step's WMMAs)
    v16bf a_nxt = cat8(*(const v8bf*)(abase + kn), *(const v8bf*)(abase + kn + 8));
    v16bf b_nxt[NT];
#pragma unroll
    for (int t = 0; t < NT; ++t)
      b_nxt[t] = cat8(*(const v8bf*)(wcol[t] + kn), *(const v8bf*)(wcol[t] + kn + 8));
    __builtin_prefetch(abase + kn + 32, 0, 1);
#pragma unroll
    for (int t = 0; t < NT; ++t) acc[t] = wmma_bf16(a_cur, b_cur[t], acc[t]);
    a_cur = a_nxt;
#pragma unroll
    for (int t = 0; t < NT; ++t) b_cur[t] = b_nxt[t];
  }
#pragma unroll
  for (int t = 0; t < NT; ++t) acc[t] = wmma_bf16(a_cur, b_cur[t], acc[t]);

#pragma unroll
  for (int t = 0; t < NT; ++t) {
    int nc = n0 + t * 16;
#pragma unroll
    for (int r = 0; r < 8; ++r) {
      int m = m0 + r + 8 * hi;
      float v = acc[t][r];
      if (act == 1)  // exact GELU
        v = 0.5f * v * (1.0f + erff(v * 0.70710678118654752f));
      if (OutF) OutF[(long)m * N + nc + (lane & 15)] = v;
      if (OutB) OutB[(long)m * N + nc + (lane & 15)] = (bf16)v;
    }
  }
}

// ---------------- RMSNorm (fp32 in, bf16 out), one wave per row -------------
__global__ void rms_kernel(const float* __restrict__ X, const float* __restrict__ Wt,
                           bf16* __restrict__ Y, int R, int Wd) {
  int wid = (blockIdx.x * blockDim.x + threadIdx.x) >> 5;
  int lane = threadIdx.x & 31;
  if (wid >= R) return;
  const float* x = X + (long)wid * Wd;
  float s = 0.f;
  for (int c = lane; c < Wd; c += 32) s += x[c] * x[c];
#pragma unroll
  for (int m = 16; m >= 1; m >>= 1) s += __shfl_xor(s, m, 32);
  float r = rsqrtf(s / (float)Wd + EPSF);
  for (int c = lane; c < Wd; c += 32) Y[(long)wid * Wd + c] = (bf16)(x[c] * r * Wt[c]);
}

// ---------------- RoPE (half-split), pre-scale (1/H for k_rope) -------------
__global__ void rope_kernel(float* __restrict__ Y, int rows, int Hn, int posmod,
                            float pre) {
  int idx = blockIdx.x * blockDim.x + threadIdx.x;
  int total = rows * Hn * (ROPED / 2);
  if (idx >= total) return;
  int i = idx % (ROPED / 2);
  int h = (idx / (ROPED / 2)) % Hn;
  int row = idx / ((ROPED / 2) * Hn);
  float* y = Y + (long)row * Hn * ROPED + h * ROPED;
  float xr = y[i] * pre, xi = y[i + ROPED / 2] * pre;
  float freq = powf(10000.f, -(float)(2 * i) / (float)ROPED);
  float ang = (float)(row % posmod) * freq;
  float cs = cosf(ang), sn = sinf(ang);
  y[i] = xr * cs - xi * sn;
  y[i + ROPED / 2] = xr * sn + xi * cs;
}

// ---------------- row dot (importance), one wave per row --------------------
__global__ void rowdot_kernel(const float* __restrict__ X, const float* __restrict__ Wv,
                              float* __restrict__ out, int R, int K) {
  int wid = (blockIdx.x * blockDim.x + threadIdx.x) >> 5;
  int lane = threadIdx.x & 31;
  if (wid >= R) return;
  float s = 0.f;
  for (int c = lane; c < K; c += 32) s += X[(long)wid * K + c] * Wv[c];
#pragma unroll
  for (int m = 16; m >= 1; m >>= 1) s += __shfl_xor(s, m, 32);
  if (lane == 0) out[wid] = s;
}

// ---------------- top-256 by rank (stable), ascending index -----------------
__global__ void topk_kernel(const float* __restrict__ imp, int* __restrict__ idx) {
  __shared__ float vals[TT];
  __shared__ int flg[TT];
  int b = blockIdx.x;
  int i = threadIdx.x;  // blockDim = TT
  vals[i] = imp[b * TT + i];
  __syncthreads();
  float vi = vals[i];
  int cnt = 0;
  for (int j = 0; j < TT; ++j) {
    float vj = vals[j];
    cnt += (vj > vi) || (vj == vi && j < i);
  }
  flg[i] = (cnt < KEEPN) ? 1 : 0;
  __syncthreads();
  if (flg[i]) {
    int pos = 0;
    for (int j = 0; j < i; ++j) pos += flg[j];
    idx[b * KEEPN + pos] = i;
  }
}

__global__ void gather_kernel(const float* __restrict__ X, const int* __restrict__ idx,
                              bf16* __restrict__ sel) {
  long t = blockIdx.x * (long)blockDim.x + threadIdx.x;
  long total = (long)BB * KEEPN * CC;
  if (t >= total) return;
  int c = t % CC;
  int r = (t / CC) % KEEPN;
  int b = t / ((long)CC * KEEPN);
  sel[t] = (bf16)X[((long)b * TT + idx[b * KEEPN + r]) * CC + c];
}

// ---------------- blocks = x.reshape + pos_enc -> bf16 ----------------------
__global__ void blocks_kernel(const float* __restrict__ X, const float* __restrict__ pe,
                              bf16* __restrict__ Ab) {
  long t = blockIdx.x * (long)blockDim.x + threadIdx.x;
  long total = (long)BB * NBLK * BLKSZ * CC;
  if (t >= total) return;
  int c = t % CC;
  int j = (t / CC) % BLKSZ;
  int nb = (t / ((long)CC * BLKSZ)) % NBLK;
  int b = t / ((long)CC * BLKSZ * NBLK);
  Ab[t] = (bf16)(X[((long)b * TT + nb * BLKSZ + j) * CC + c] + pe[j * CC + c]);
}

// ---------------- q = concat(q_nope, q_rope) -> bf16 [B*T, H*96] ------------
__global__ void buildq_kernel(const float* __restrict__ qn, const float* __restrict__ qr,
                              bf16* __restrict__ Q) {
  long t = blockIdx.x * (long)blockDim.x + threadIdx.x;
  long total = (long)BB * TT * HH * QKD;
  if (t >= total) return;
  int d = t % QKD;
  int h = (t / QKD) % HH;
  long row = t / ((long)QKD * HH);
  float v = (d < NOPE) ? qn[row * (HH * NOPE) + h * NOPE + d]
                       : qr[row * (HH * ROPED) + h * ROPED + (d - NOPE)];
  Q[row * (HH * QKD) + h * QKD + d] = (bf16)v;
}

// ---------------- k1 = concat(k_nope, bcast k_rope) -> bf16 [B*NB, H*96] ----
__global__ void buildk1_kernel(const float* __restrict__ kn, const float* __restrict__ kr,
                               bf16* __restrict__ K1) {
  long t = blockIdx.x * (long)blockDim.x + threadIdx.x;
  long total = (long)BB * NBLK * HH * QKD;
  if (t >= total) return;
  int d = t % QKD;
  int h = (t / QKD) % HH;
  long row = t / ((long)QKD * HH);
  float v = (d < NOPE) ? kn[row * (HH * NOPE) + h * NOPE + d]
                       : kr[row * ROPED + (d - NOPE)];
  K1[row * (HH * QKD) + h * QKD + d] = (bf16)v;
}

// ---------------- V fp32 [B*S, H*32] -> V^T bf16 [B,H,32,S] -----------------
__global__ void vtrans_kernel(const float* __restrict__ V, bf16* __restrict__ VT,
                              int S) {
  long t = blockIdx.x * (long)blockDim.x + threadIdx.x;
  long total = (long)BB * HH * VDIM * S;
  if (t >= total) return;
  int s = t % S;
  int vd = (t / S) % VDIM;
  int h = (t / ((long)S * VDIM)) % HH;
  int b = t / ((long)S * VDIM * HH);
  VT[t] = (bf16)V[((long)b * S + s) * (HH * VDIM) + h * VDIM + vd];
}

// ---------------- WMMA attention, one wave per 16-query tile per (b,h) ------
// Q: bf16 [B*T, H*96]; Km: bf16 [B*S, H*96]; VT: bf16 [B,H,32,S]; O: [B,H,T,32].
__global__ void attn_kernel(const bf16* __restrict__ Q, const bf16* __restrict__ Km,
                            const bf16* __restrict__ VT, float* __restrict__ O,
                            int S, int windowed) {
  __shared__ float sc[16][17 * 16];   // raw scores / softmax workspace
  __shared__ bf16 pb[16][17 * 16];    // normalized probs, bf16, b128-aligned rows
  const int lane = threadIdx.x & 31;
  const int hi = lane >> 4;
  const int t0 = blockIdx.x * 16;
  const int h = blockIdx.y;
  const int b = blockIdx.z;
  int startkey, ntiles;
  if (windowed) {
    startkey = (t0 >= WINSZ) ? (t0 - WINSZ) : 0;
    ntiles = (t0 + 16 - startkey) >> 4;
  } else {
    startkey = 0;
    ntiles = S >> 4;
  }
  // preload 3 Q fragments (d = 0,32,64), all b128 loads
  const bf16* qbase =
      Q + ((long)b * TT + t0 + (lane & 15)) * (HH * QKD) + h * QKD + hi * 16;
  v16bf qf[3];
#pragma unroll
  for (int ds = 0; ds < 3; ++ds)
    qf[ds] = cat8(*(const v8bf*)(qbase + ds * 32),
                  *(const v8bf*)(qbase + ds * 32 + 8));
  // scores = Q K^T -> LDS (all 6 loads clause before the WMMA chain)
  for (int kt = 0; kt < ntiles; ++kt) {
    int key0 = startkey + kt * 16;
    const bf16* kbase =
        Km + ((long)b * S + key0 + (lane & 15)) * (HH * QKD) + h * QKD + hi * 16;
    v16bf kf0 = cat8(*(const v8bf*)(kbase), *(const v8bf*)(kbase + 8));
    v16bf kf1 = cat8(*(const v8bf*)(kbase + 32), *(const v8bf*)(kbase + 40));
    v16bf kf2 = cat8(*(const v8bf*)(kbase + 64), *(const v8bf*)(kbase + 72));
    v8f acc = {};
    acc = wmma_bf16(qf[0], kf0, acc);
    acc = wmma_bf16(qf[1], kf1, acc);
    acc = wmma_bf16(qf[2], kf2, acc);
#pragma unroll
    for (int r = 0; r < 8; ++r)
      sc[r + 8 * hi][kt * 16 + (lane & 15)] = acc[r];
  }
  __syncthreads();
  // row softmax (lane l < 16 owns query row t0+l); write bf16 probs
  int ncols = ntiles * 16;
  if (lane < 16) {
    int ig = t0 + lane;
    float mx = -3.0e38f;
    for (int c = 0; c < ncols; ++c) {
      float s = sc[lane][c] * SCALEF;
      if (windowed) {
        int j = startkey + c;
        if (!(j <= ig && (ig - j) < WINSZ)) s = -1.0e30f;
      }
      sc[lane][c] = s;
      mx = fmaxf(mx, s);
    }
    float sum = 0.f;
    for (int c = 0; c < ncols; ++c) {
      float e = __expf(sc[lane][c] - mx);
      sc[lane][c] = e;
      sum += e;
    }
    float inv = 1.f / sum;
    for (int c = 0; c < ncols; ++c) pb[lane][c] = (bf16)(sc[lane][c] * inv);
  }
  __syncthreads();
  // O = P V  (P from LDS as b128, V^T rows contiguous along keys)
  const bf16* pbase = &pb[lane & 15][0] + hi * 16;
  const bf16* vbase0 =
      VT + ((long)(b * HH + h) * VDIM + (lane & 15)) * S + startkey + hi * 16;
  const bf16* vbase1 = vbase0 + (long)16 * S;
  v8f oacc0 = {}, oacc1 = {};
  for (int kt = 0; kt < ntiles; ++kt) {
    v16bf pa = cat8(*(const v8bf*)(pbase + kt * 16),
                    *(const v8bf*)(pbase + kt * 16 + 8));
    v16bf vb0 = cat8(*(const v8bf*)(vbase0 + kt * 16),
                     *(const v8bf*)(vbase0 + kt * 16 + 8));
    v16bf vb1 = cat8(*(const v8bf*)(vbase1 + kt * 16),
                     *(const v8bf*)(vbase1 + kt * 16 + 8));
    oacc0 = wmma_bf16(pa, vb0, oacc0);
    oacc1 = wmma_bf16(pa, vb1, oacc1);
  }
  float* orow = O + ((long)(b * HH + h) * TT + t0) * VDIM;
#pragma unroll
  for (int r = 0; r < 8; ++r) {
    int m = r + 8 * hi;
    orow[(long)m * VDIM + (lane & 15)] = oacc0[r];
    orow[(long)m * VDIM + 16 + (lane & 15)] = oacc1[r];
  }
}

// ---------------- gate softmax over 3 logits (one wave per token) -----------
__global__ void gate_kernel(const float* __restrict__ X, const float* __restrict__ Wg,
                            float* __restrict__ G, int R) {
  int wid = (blockIdx.x * blockDim.x + threadIdx.x) >> 5;
  int lane = threadIdx.x & 31;
  if (wid >= R) return;
  float s[3];
#pragma unroll
  for (int g = 0; g < 3; ++g) {
    float acc = 0.f;
    for (int c = lane; c < CC; c += 32) acc += X[(long)wid * CC + c] * Wg[c * 3 + g];
#pragma unroll
    for (int m = 16; m >= 1; m >>= 1) acc += __shfl_xor(acc, m, 32);
    s[g] = acc;
  }
  if (lane == 0) {
    float mx = fmaxf(s[0], fmaxf(s[1], s[2]));
    float e0 = __expf(s[0] - mx), e1 = __expf(s[1] - mx), e2 = __expf(s[2] - mx);
    float inv = 1.f / (e0 + e1 + e2);
    G[wid * 3 + 0] = e0 * inv;
    G[wid * 3 + 1] = e1 * inv;
    G[wid * 3 + 2] = e2 * inv;
  }
}

// ---------------- gated combine + head transpose -> bf16 [B*T, H*32] --------
__global__ void combine_kernel(const float* __restrict__ o1, const float* __restrict__ o2,
                               const float* __restrict__ o3, const float* __restrict__ G,
                               bf16* __restrict__ comb) {
  long t = blockIdx.x * (long)blockDim.x + threadIdx.x;
  long total = (long)BB * TT * HH * VDIM;
  if (t >= total) return;
  int vd = t % VDIM;
  int h = (t / VDIM) % HH;
  int tt = (t / ((long)VDIM * HH)) % TT;
  int b = t / ((long)VDIM * HH * TT);
  long orow = ((long)(b * HH + h) * TT + tt) * VDIM + vd;
  long g = (long)(b * TT + tt) * 3;
  comb[((long)(b * TT + tt) * HH + h) * VDIM + vd] =
      (bf16)(G[g] * o1[orow] + G[g + 1] * o2[orow] + G[g + 2] * o3[orow]);
}

// ---------------- host orchestration ----------------------------------------
extern "C" void kernel_launch(void* const* d_in, const int* in_sizes, int n_in,
                              void* d_out, int out_size, void* d_ws, size_t ws_size,
                              hipStream_t stream) {
  const float* x     = (const float*)d_in[0];
  const float* Wcq   = (const float*)d_in[1];
  const float* qnw   = (const float*)d_in[2];
  const float* Wdqn  = (const float*)d_in[3];
  const float* Wdqr  = (const float*)d_in[4];
  const float* Wckv  = (const float*)d_in[5];
  const float* kvnw  = (const float*)d_in[6];
  const float* Wdkn  = (const float*)d_in[7];
  const float* Wdv   = (const float*)d_in[8];
  const float* Wkr   = (const float*)d_in[9];
  const float* Wimp  = (const float*)d_in[10];
  const float* Wselk = (const float*)d_in[11];
  const float* Wselv = (const float*)d_in[12];
  const float* Wwink = (const float*)d_in[13];
  const float* Wwinv = (const float*)d_in[14];
  const float* Wc1   = (const float*)d_in[15];
  const float* Wc2   = (const float*)d_in[16];
  const float* pe    = (const float*)d_in[17];
  const float* Wgate = (const float*)d_in[18];
  const float* Wproj = (const float*)d_in[19];
  float* out = (float*)d_out;

  // byte bump allocator over d_ws (~215 MB: bf16 weight copies dominate)
  char* base = (char*)d_ws;
  size_t off = 0;
  auto allocB = [&](size_t bytes) -> void* {
    void* p = (void*)(base + off);
    off = (off + bytes + 255) & ~(size_t)255;
    return p;
  };
  auto allocF = [&](size_t n) -> float* { return (float*)allocB(n * 4); };
  auto allocH = [&](size_t n) -> bf16* { return (bf16*)allocB(n * 2); };

  const int NTOK = BB * TT;  // 2048
  // transposed bf16 weights
  bf16* WcqT   = allocH((long)CC * QLORA);
  bf16* WdqnT  = allocH((long)QLORA * HH * NOPE);
  bf16* WdqrT  = allocH((long)QLORA * HH * ROPED);
  bf16* WckvT  = allocH((long)CC * KVLORA);
  bf16* WdknT  = allocH((long)KVLORA * HH * NOPE);
  bf16* WdvT   = allocH((long)KVLORA * HH * VDIM);
  bf16* WkrT   = allocH((long)CC * ROPED);
  bf16* WselkT = allocH((long)CC * HH * QKD);
  bf16* WselvT = allocH((long)CC * HH * VDIM);
  bf16* WwinkT = allocH((long)CC * HH * QKD);
  bf16* WwinvT = allocH((long)CC * HH * VDIM);
  bf16* Wc1T   = allocH((long)BLKSZ * CC * 4 * CC);
  bf16* Wc2T   = allocH((long)4 * CC * CC);
  bf16* WprojT = allocH((long)HH * VDIM * CC);
  // activations
  bf16*  xbf     = allocH((long)NTOK * CC);
  float* nq_pre  = allocF((long)NTOK * QLORA);
  bf16*  nqb     = allocH((long)NTOK * QLORA);
  float* q_nope  = allocF((long)NTOK * HH * NOPE);
  float* q_rope  = allocF((long)NTOK * HH * ROPED);
  bf16*  qallb   = allocH((long)NTOK * HH * QKD);
  bf16*  blocksb = allocH((long)BB * NBLK * BLKSZ * CC);
  bf16*  cb1b    = allocH((long)BB * NBLK * 4 * CC);
  bf16*  cbb     = allocH((long)BB * NBLK * CC);
  float* nkv_pre = allocF((long)BB * NBLK * KVLORA);
  bf16*  nkvb    = allocH((long)BB * NBLK * KVLORA);
  float* k_nope  = allocF((long)BB * NBLK * HH * NOPE);
  float* v1      = allocF((long)BB * NBLK * HH * VDIM);
  float* k_rope  = allocF((long)BB * NBLK * ROPED);
  bf16*  k1b     = allocH((long)BB * NBLK * HH * QKD);
  bf16*  vt1     = allocH((long)BB * HH * VDIM * NBLK);
  float* impv    = allocF(NTOK);
  int*   idx     = (int*)allocB(BB * KEEPN * 4);
  bf16*  selb    = allocH((long)BB * KEEPN * CC);
  bf16*  selkb   = allocH((long)BB * KEEPN * HH * QKD);
  float* selv    = allocF((long)BB * KEEPN * HH * VDIM);
  bf16*  vt2     = allocH((long)BB * HH * VDIM * KEEPN);
  bf16*  winkb   = allocH((long)NTOK * HH * QKD);
  float* winv    = allocF((long)NTOK * HH * VDIM);
  bf16*  vt3     = allocH((long)BB * HH * VDIM * TT);
  float* o1      = allocF((long)NTOK * HH * VDIM);
  float* o2      = allocF((long)NTOK * HH * VDIM);
  float* o3      = allocF((long)NTOK * HH * VDIM);
  float* gate    = allocF((long)NTOK * 3);
  bf16*  combb   = allocH((long)NTOK * HH * VDIM);

  auto wtrans = [&](const float* W, bf16* WT, int K, int N) {
    dim3 g((N + 31) / 32, (K + 31) / 32);
    wtrans_kernel<<<g, 256, 0, stream>>>(W, WT, K, N);
  };
  // NT chosen so N % (16*NT) == 0 and grid has enough blocks.
  auto gemm = [&](const bf16* A, const bf16* WT, float* OF, bf16* OB, int M, int N,
                  int K, int NT, int act) {
    dim3 g(N / (16 * NT), M / 64);
    if (NT == 1)
      gemm_bf16_kernel<1><<<g, 128, 0, stream>>>(A, WT, OF, OB, M, N, K, act);
    else if (NT == 2)
      gemm_bf16_kernel<2><<<g, 128, 0, stream>>>(A, WT, OF, OB, M, N, K, act);
    else
      gemm_bf16_kernel<4><<<g, 128, 0, stream>>>(A, WT, OF, OB, M, N, K, act);
  };

  // --- weight transforms (bf16, column-major) ---
  wtrans(Wcq, WcqT, CC, QLORA);
  wtrans(Wdqn, WdqnT, QLORA, HH * NOPE);
  wtrans(Wdqr, WdqrT, QLORA, HH * ROPED);
  wtrans(Wckv, WckvT, CC, KVLORA);
  wtrans(Wdkn, WdknT, KVLORA, HH * NOPE);
  wtrans(Wdv, WdvT, KVLORA, HH * VDIM);
  wtrans(Wkr, WkrT, CC, ROPED);
  wtrans(Wselk, WselkT, CC, HH * QKD);
  wtrans(Wselv, WselvT, CC, HH * VDIM);
  wtrans(Wwink, WwinkT, CC, HH * QKD);
  wtrans(Wwinv, WwinvT, CC, HH * VDIM);
  wtrans(Wc1, Wc1T, BLKSZ * CC, 4 * CC);
  wtrans(Wc2, Wc2T, 4 * CC, CC);
  wtrans(Wproj, WprojT, HH * VDIM, CC);
  {
    long n = (long)NTOK * CC;
    castbf_kernel<<<(int)((n + 255) / 256), 256, 0, stream>>>(x, xbf, n);
  }

  // --- MLA queries ---
  gemm(xbf, WcqT, nq_pre, nullptr, NTOK, QLORA, CC, 2, 0);
  rms_kernel<<<(NTOK * 32 + 127) / 128, 128, 0, stream>>>(nq_pre, qnw, nqb, NTOK, QLORA);
  gemm(nqb, WdqnT, q_nope, nullptr, NTOK, HH * NOPE, QLORA, 4, 0);
  gemm(nqb, WdqrT, q_rope, nullptr, NTOK, HH * ROPED, QLORA, 4, 0);
  {
    int tot = NTOK * HH * (ROPED / 2);
    rope_kernel<<<(tot + 255) / 256, 256, 0, stream>>>(q_rope, NTOK, HH, TT, 1.0f);
  }
  {
    long tot = (long)NTOK * HH * QKD;
    buildq_kernel<<<(int)((tot + 255) / 256), 256, 0, stream>>>(q_nope, q_rope, qallb);
  }

  // --- branch 1: block compression + MLA KV ---
  {
    long tot = (long)BB * NBLK * BLKSZ * CC;
    blocks_kernel<<<(int)((tot + 255) / 256), 256, 0, stream>>>(x, pe, blocksb);
  }
  // small-M GEMMs: narrow NT for block-level parallelism
  gemm(blocksb, Wc1T, nullptr, cb1b, BB * NBLK, 4 * CC, BLKSZ * CC, 2, 1);  // GELU
  gemm(cb1b, Wc2T, nullptr, cbb, BB * NBLK, CC, 4 * CC, 1, 0);
  gemm(cbb, WckvT, nkv_pre, nullptr, BB * NBLK, KVLORA, CC, 2, 0);
  rms_kernel<<<(BB * NBLK * 32 + 127) / 128, 128, 0, stream>>>(nkv_pre, kvnw, nkvb,
                                                              BB * NBLK, KVLORA);
  gemm(nkvb, WdknT, k_nope, nullptr, BB * NBLK, HH * NOPE, KVLORA, 4, 0);
  gemm(nkvb, WdvT, v1, nullptr, BB * NBLK, HH * VDIM, KVLORA, 4, 0);
  gemm(cbb, WkrT, k_rope, nullptr, BB * NBLK, ROPED, CC, 1, 0);
  {
    int tot = BB * NBLK * (ROPED / 2);
    rope_kernel<<<(tot + 255) / 256, 256, 0, stream>>>(k_rope, BB * NBLK, 1, NBLK,
                                                       1.0f / (float)HH);
  }
  {
    long tot = (long)BB * NBLK * HH * QKD;
    buildk1_kernel<<<(int)((tot + 255) / 256), 256, 0, stream>>>(k_nope, k_rope, k1b);
  }
  {
    long tot = (long)BB * HH * VDIM * NBLK;
    vtrans_kernel<<<(int)((tot + 255) / 256), 256, 0, stream>>>(v1, vt1, NBLK);
  }
  attn_kernel<<<dim3(TT / 16, HH, BB), 32, 0, stream>>>(qallb, k1b, vt1, o1, NBLK, 0);

  // --- branch 2: top-k selection ---
  rowdot_kernel<<<(NTOK * 32 + 127) / 128, 128, 0, stream>>>(x, Wimp, impv, NTOK, CC);
  topk_kernel<<<BB, TT, 0, stream>>>(impv, idx);
  {
    long tot = (long)BB * KEEPN * CC;
    gather_kernel<<<(int)((tot + 255) / 256), 256, 0, stream>>>(x, idx, selb);
  }
  gemm(selb, WselkT, nullptr, selkb, BB * KEEPN, HH * QKD, CC, 4, 0);
  gemm(selb, WselvT, selv, nullptr, BB * KEEPN, HH * VDIM, CC, 2, 0);
  {
    long tot = (long)BB * HH * VDIM * KEEPN;
    vtrans_kernel<<<(int)((tot + 255) / 256), 256, 0, stream>>>(selv, vt2, KEEPN);
  }
  attn_kernel<<<dim3(TT / 16, HH, BB), 32, 0, stream>>>(qallb, selkb, vt2, o2, KEEPN, 0);

  // --- branch 3: sliding window ---
  gemm(xbf, WwinkT, nullptr, winkb, NTOK, HH * QKD, CC, 4, 0);
  gemm(xbf, WwinvT, winv, nullptr, NTOK, HH * VDIM, CC, 4, 0);
  {
    long tot = (long)BB * HH * VDIM * TT;
    vtrans_kernel<<<(int)((tot + 255) / 256), 256, 0, stream>>>(winv, vt3, TT);
  }
  attn_kernel<<<dim3(TT / 16, HH, BB), 32, 0, stream>>>(qallb, winkb, vt3, o3, TT, 1);

  // --- gate, combine, project ---
  gate_kernel<<<(NTOK * 32 + 127) / 128, 128, 0, stream>>>(x, Wgate, gate, NTOK);
  {
    long tot = (long)NTOK * HH * VDIM;
    combine_kernel<<<(int)((tot + 255) / 256), 256, 0, stream>>>(o1, o2, o3, gate, combb);
  }
  gemm(combb, WprojT, out, nullptr, NTOK, CC, HH * VDIM, 4, 0);
}
